// AttLayer_11020886081977
// MI455X (gfx1250) — matrix operand order repaired and checked by reading, not compiled, not run on hardware
//
#include <hip/hip_runtime.h>
#include <hip/hip_bf16.h>

typedef __attribute__((ext_vector_type(16))) _Float16 v16h;
typedef __attribute__((ext_vector_type(8)))  _Float16 v8h;
typedef __attribute__((ext_vector_type(8)))  float    v8f;

__device__ __forceinline__ int lane_id() { return threadIdx.x & 31; }

// A-fragment (16x32 MxK, f16): lane holds row m=lane&15; lanes 0-15: K=0..15;
// lanes 16-31: K=16..31.  p = &A[row(lane)][k0], row-major, contiguous K.
__device__ __forceinline__ v16h load_a_frag(const _Float16* p) {
  int kb = (lane_id() >> 4) << 4;
  v8h lo = *(const v8h*)(p + kb);
  v8h hi = *(const v8h*)(p + kb + 8);
  v16h r;
#pragma unroll
  for (int i = 0; i < 8; ++i) { r[i] = lo[i]; r[i + 8] = hi[i]; }
  return r;
}

// B-fragment (32x16 KxN, f16): lane holds col n=lane&15; lanes 0-15: K=0..15;
// lanes 16-31: K=16..31.  p = &BT[n(lane)][k0] where BT is NxK row-major.
__device__ __forceinline__ v16h load_b_frag(const _Float16* p) {
  int kb = (lane_id() >> 4) << 4;
  v8h lo = *(const v8h*)(p + kb);
  v8h hi = *(const v8h*)(p + kb + 8);
  v16h r;
#pragma unroll
  for (int i = 0; i < 8; ++i) { r[i] = lo[i]; r[i + 8] = hi[i]; }
  return r;
}

// Double-buffered 16x16 tile GEMM over K: next-step fragment loads are issued
// before the current WMMA so the wait is on older loads only.
template<int K, int LDA, int LDB>
__device__ __forceinline__ v8f gemm_rt(const _Float16* arow, const _Float16* btrow) {
  v8f acc = {};
  v16h a0 = load_a_frag(arow);
  v16h b0 = load_b_frag(btrow);
#pragma unroll
  for (int k0 = 0; k0 < K; k0 += 32) {
    v16h a1 = {};
    v16h b1 = {};
    if (k0 + 32 < K) {
      a1 = load_a_frag(arow + k0 + 32);
      b1 = load_b_frag(btrow + k0 + 32);
    }
    acc = __builtin_amdgcn_wmma_f32_16x16x32_f16(false, a0, false, b0, (short)0, acc,
                                                 false, false);
    a0 = a1; b0 = b1;
  }
  return acc;
}

// ---------------------------------------------------------------------------
// Kernel 0: convert all four weight matrices (each 131072 elems) f32 -> f16
__global__ void __launch_bounds__(256) prep_w_kernel(
    const float* __restrict__ Wq, const float* __restrict__ Wk,
    const float* __restrict__ Wv, const float* __restrict__ Wo,
    _Float16* __restrict__ Wq16, _Float16* __restrict__ Wk16,
    _Float16* __restrict__ Wv16, _Float16* __restrict__ Wo16)
{
  int i = blockIdx.x * 256 + threadIdx.x;
  if (i < 256 * 512) {
    Wq16[i] = (_Float16)Wq[i];
    Wk16[i] = (_Float16)Wk[i];
    Wv16[i] = (_Float16)Wv[i];
    Wo16[i] = (_Float16)Wo[i];
  }
}

// ---------------------------------------------------------------------------
// Kernel 1: QKV projections, transpose+convert of x1 fused via LDS staging.
//  LDS tile xs[l][c] (64 x 520 f16, padded rows -> conflict-free ds_load_b128)
//  qT[b][l][o], kT[b][l][o]  (D rows = l, cols = o; A = xs rows, B = W rows)
//  v [b][o][l]               (D rows = o, cols = l; A = W rows, B = xs rows)
__global__ void __launch_bounds__(256) qkv_kernel(
    const float* __restrict__ x1,
    const _Float16* __restrict__ Wq16, const _Float16* __restrict__ Wk16,
    const _Float16* __restrict__ Wv16,
    const float* __restrict__ bq, const float* __restrict__ bk,
    const float* __restrict__ bv,
    _Float16* __restrict__ qT, _Float16* __restrict__ kT,
    _Float16* __restrict__ vv, int L)
{
  constexpr int XS_LD = 520;            // 512 + 8 pad
  extern __shared__ _Float16 xs[];      // 64 * 520 f16 = 66,560 B
  int b    = blockIdx.y;
  int l0   = blockIdx.x * 64;
  int tid  = threadIdx.x;
  int wave = tid >> 5;
  int lane = tid & 31;
  int rsel = lane & 15;
  int mh   = (lane >> 4) << 3;

  // Stage + transpose + convert: xs[l][c] = (f16) x1[b][c][l0+l]
  for (int i = tid; i < 64 * 512; i += 256) {
    int l = i & 63, c = i >> 6;
    const float* src = x1 + ((size_t)b * 512 + c) * L + l0 + l;
    xs[l * XS_LD + c] = (_Float16)(*src);
    __builtin_prefetch(src + 64, 0, 1);   // next block's stripe -> global_prefetch
  }
  __syncthreads();

  for (int t8 = 0; t8 < 8; ++t8) {
    int t = wave * 8 + t8;
    {
      int mt = t >> 4;   // 0..3  (l tile)
      int nt = t & 15;   // 0..15 (o tile)
      const _Float16* arow = xs + (size_t)(mt * 16 + rsel) * XS_LD;
      {
        v8f acc = gemm_rt<512, XS_LD, 512>(arow, Wq16 + (size_t)(nt * 16 + rsel) * 512);
        float bias = bq[nt * 16 + rsel];
        _Float16* d = qT + ((size_t)b * L + l0 + mt * 16) * 256 + nt * 16;
#pragma unroll
        for (int r = 0; r < 8; ++r)
          d[(size_t)(mh + r) * 256 + rsel] = (_Float16)(acc[r] + bias);
      }
      {
        v8f acc = gemm_rt<512, XS_LD, 512>(arow, Wk16 + (size_t)(nt * 16 + rsel) * 512);
        float bias = bk[nt * 16 + rsel];
        _Float16* d = kT + ((size_t)b * L + l0 + mt * 16) * 256 + nt * 16;
#pragma unroll
        for (int r = 0; r < 8; ++r)
          d[(size_t)(mh + r) * 256 + rsel] = (_Float16)(acc[r] + bias);
      }
    }
    {
      int mt = t >> 2;   // 0..15 (o tile)
      int nt = t & 3;    // 0..3  (l tile)
      const _Float16* arow = Wv16 + (size_t)(mt * 16 + rsel) * 512;
      const _Float16* brow = xs + (size_t)(nt * 16 + rsel) * XS_LD;
      v8f acc = gemm_rt<512, 512, XS_LD>(arow, brow);
      _Float16* d = vv + ((size_t)b * 256 + mt * 16) * L + l0 + nt * 16;
#pragma unroll
      for (int r = 0; r < 8; ++r) {
        float bias = bv[mt * 16 + mh + r];
        d[(size_t)(mh + r) * L + rsel] = (_Float16)(acc[r] + bias);
      }
    }
  }
}

// ---------------------------------------------------------------------------
// Kernel 2: sliding-window attention. One block per (batch, 512-query block).
// 8 waves; each wave handles 64 query rows as 4 tiles of 16, keeping its
// 16x1024 probability panel in its private 32KB LDS slice (256KB total).
__global__ void __launch_bounds__(256) attn_kernel(
    const _Float16* __restrict__ qT, const _Float16* __restrict__ kT,
    const _Float16* __restrict__ vv, const float* __restrict__ mask,
    _Float16* __restrict__ aT, int L)
{
  extern __shared__ _Float16 satt[];   // 8 * 16 * 1024 f16 = 256 KB
  int b    = blockIdx.y;
  int l0   = blockIdx.x * 512;
  int w0   = l0 - 256;                 // window start (may be negative)
  int wave = threadIdx.x >> 5;
  int lane = threadIdx.x & 31;
  int rsel = lane & 15;
  int mh   = (lane >> 4) << 3;
  _Float16* my = satt + (size_t)wave * 16 * 1024;
  const _Float16* qTb = qT + (size_t)b * L * 256;
  const _Float16* kTb = kT + (size_t)b * L * 256;
  const _Float16* vb  = vv + (size_t)b * 256 * L;

  for (int qt = 0; qt < 4; ++qt) {
    int q0 = wave * 64 + qt * 16;
    const _Float16* arow = qTb + (size_t)(l0 + q0 + rsel) * 256;
    float rs[8];
#pragma unroll
    for (int r = 0; r < 8; ++r) rs[r] = 0.f;

    // Pass A: energy -> exp -> LDS (unnormalized, mask-multiplied)
    for (int nt = 0; nt < 64; ++nt) {
      int  kp  = w0 + nt * 16 + rsel;       // key position for this lane's column
      bool kok = (kp >= 0) && (kp < L);
      const _Float16* brow = kTb + (size_t)(kok ? kp : 0) * 256;
      v8f acc = {};
#pragma unroll
      for (int k0 = 0; k0 < 256; k0 += 32) {
        v16h a = load_a_frag(arow + k0);
        v16h bf;
        if (kok) bf = load_b_frag(brow + k0);
        else     { v16h z = {}; bf = z; }    // zero-padded keys (exact)
        acc = __builtin_amdgcn_wmma_f32_16x16x32_f16(false, a, false, bf, (short)0,
                                                     acc, false, false);
      }
      float mv = kok ? mask[(size_t)b * L + kp] : 0.f;
      float lm = __logf(mv + 1e-6f);
#pragma unroll
      for (int r = 0; r < 8; ++r) {
        float pn = __expf(acc[r] * 0.0625f + lm);   // 1/sqrt(256) = 1/16
        rs[r] += pn;                                 // softmax denominator
        my[(size_t)(mh + r) * 1024 + nt * 16 + rsel] = (_Float16)(pn * mv);
      }
    }
    // row-sum reduce across the 16 lanes sharing each row half
#pragma unroll
    for (int r = 0; r < 8; ++r) {
      rs[r] += __shfl_xor(rs[r], 1, 32);
      rs[r] += __shfl_xor(rs[r], 2, 32);
      rs[r] += __shfl_xor(rs[r], 4, 32);
      rs[r] += __shfl_xor(rs[r], 8, 32);
    }
    float inv[8];
#pragma unroll
    for (int r = 0; r < 8; ++r) inv[r] = 1.0f / rs[r];

    asm volatile("s_wait_dscnt 0" ::: "memory");

    // Pass B: out = att * V  (A from LDS panel, B from v rows; OOB l clamped
    // — boundaries are 256-aligned so clamped runs always carry att == 0)
    for (int ct = 0; ct < 16; ++ct) {
      const _Float16* arow2 = my + (size_t)rsel * 1024;
      const _Float16* vrow  = vb + (size_t)(ct * 16 + rsel) * L;
      int kbB = (lane >> 4) << 4;
      v8f oacc = {};
      for (int k2 = 0; k2 < 1024; k2 += 32) {
        v16h a = load_a_frag(arow2 + k2);
        int lb  = w0 + k2 + kbB;
        int lb0 = lb;     if (lb0 < 0) lb0 = 0; if (lb0 > L - 8) lb0 = L - 8;
        int lb1 = lb + 8; if (lb1 < 0) lb1 = 0; if (lb1 > L - 8) lb1 = L - 8;
        v8h blo = *(const v8h*)(vrow + lb0);
        v8h bhi = *(const v8h*)(vrow + lb1);
        v16h bf;
#pragma unroll
        for (int i = 0; i < 8; ++i) { bf[i] = blo[i]; bf[i + 8] = bhi[i]; }
        oacc = __builtin_amdgcn_wmma_f32_16x16x32_f16(false, a, false, bf, (short)0,
                                                      oacc, false, false);
      }
#pragma unroll
      for (int r = 0; r < 8; ++r) {
        float val = oacc[r] * inv[r];
        val = val > 0.f ? val : 0.f;                 // fused ReLU
        aT[((size_t)b * L + l0 + q0 + mh + r) * 256 + ct * 16 + rsel] = (_Float16)val;
      }
    }
    asm volatile("s_wait_dscnt 0" ::: "memory");
  }
}

// ---------------------------------------------------------------------------
// Kernel 3: output projection y[b][o][l] = (Wo * relu(att_out))[o][l] + bo, * mask
// aT tile staged into LDS once via CDNA5 async global->LDS copies (ASYNCcnt).
__global__ void __launch_bounds__(256) outproj_kernel(
    const _Float16* __restrict__ aT, const _Float16* __restrict__ Wo16,
    const float* __restrict__ bo, const float* __restrict__ mask,
    float* __restrict__ y, int L)
{
  constexpr int AS_LD = 264;            // 256 + 8 pad
  __shared__ _Float16 as_[64 * AS_LD];  // 33,792 B, allocated at LDS offset 0
  int b    = blockIdx.y;
  int l0   = blockIdx.x * 64;
  int tid  = threadIdx.x;
  int wave = tid >> 5;
  int lane = tid & 31;
  int rsel = lane & 15;
  int mh   = (lane >> 4) << 3;

  // Async-stage aT rows: as_[l][0..255] = aT[b][l0+l][0..255], 16B chunks
  {
    const _Float16* srcbase = aT + ((size_t)b * L + l0) * 256;
    for (int i = tid; i < 64 * 32; i += 256) {   // 2048 x 16-byte chunks
      int l = i >> 5, cc = i & 31;
      const _Float16* g = srcbase + (size_t)l * 256 + cc * 8;
      int ldsoff = (l * AS_LD + cc * 8) * (int)sizeof(_Float16);
      asm volatile("global_load_async_to_lds_b128 %0, %1, off"
                   :: "v"(ldsoff), "v"(g) : "memory");
    }
    asm volatile("s_wait_asynccnt 0" ::: "memory");
  }
  __syncthreads();

  for (int i = 0; i < 16; ++i) {
    int t  = wave * 16 + i;
    int mt = t >> 2;   // 0..31 (o tile)
    int nt = t & 3;    // 0..3  (l tile)
    const _Float16* arow = Wo16 + (size_t)(mt * 16 + rsel) * 256;
    const _Float16* brow = as_ + (size_t)(nt * 16 + rsel) * AS_LD;
    v8f acc = gemm_rt<256, 256, AS_LD>(arow, brow);
    float mval = mask[(size_t)b * L + l0 + nt * 16 + rsel];
    float* d = y + ((size_t)b * 512 + mt * 16) * L + l0 + nt * 16;
#pragma unroll
    for (int r = 0; r < 8; ++r) {
      float bias = bo[mt * 16 + mh + r];
      d[(size_t)(mh + r) * L + rsel] = (acc[r] + bias) * mval;
    }
  }
}

// ---------------------------------------------------------------------------
extern "C" void kernel_launch(void* const* d_in, const int* in_sizes, int n_in,
                              void* d_out, int out_size, void* d_ws, size_t ws_size,
                              hipStream_t stream)
{
  const int B = 4, L = 8192;
  const float* x1   = (const float*)d_in[0];
  const float* mask = (const float*)d_in[1];
  const float* Wq   = (const float*)d_in[2];
  const float* bq   = (const float*)d_in[3];
  const float* Wk   = (const float*)d_in[4];
  const float* bk   = (const float*)d_in[5];
  const float* Wv   = (const float*)d_in[6];
  const float* bv   = (const float*)d_in[7];
  const float* Wo   = (const float*)d_in[8];
  const float* bo   = (const float*)d_in[9];
  float* y = (float*)d_out;

  _Float16* p    = (_Float16*)d_ws;
  _Float16* qT   = p; p += (size_t)B * L * 256;
  _Float16* kT   = p; p += (size_t)B * L * 256;
  _Float16* vv   = p; p += (size_t)B * L * 256;
  _Float16* aT   = p; p += (size_t)B * L * 256;
  _Float16* Wq16 = p; p += 256 * 512;
  _Float16* Wk16 = p; p += 256 * 512;
  _Float16* Wv16 = p; p += 256 * 512;
  _Float16* Wo16 = p; p += 512 * 256;

  prep_w_kernel<<<512, 256, 0, stream>>>(Wq, Wk, Wv, Wo, Wq16, Wk16, Wv16, Wo16);
  qkv_kernel<<<dim3(L / 64, B), 256, 64 * 520 * sizeof(_Float16), stream>>>(
      x1, Wq16, Wk16, Wv16, bq, bk, bv, qT, kT, vv, L);
  attn_kernel<<<dim3(L / 512, B), 256, 8 * 16 * 1024 * sizeof(_Float16), stream>>>(
      qT, kT, vv, mask, aT, L);
  outproj_kernel<<<dim3(L / 64, B), 256, 0, stream>>>(aT, Wo16, bo, mask, y, L);
}
